// PathEmbedding_1752346656912
// MI455X (gfx1250) — matrix operand order repaired and checked by reading, not compiled, not run on hardware
//
#include <hip/hip_runtime.h>

#define NUM_EDGES 10000
#define NUM_PATHS 30000
#define MAX_LEN   16
#define LINK_DIM  128
#define PATH_DIM  128

typedef __bf16 bf16_t;
typedef __attribute__((ext_vector_type(16))) __bf16 bf16x16;
typedef __attribute__((ext_vector_type(8)))  __bf16 bf16x8;
typedef __attribute__((ext_vector_type(8)))  float  f32x8;

// Workgroup = 4 waves cooperating on ONE 16-path tile.
// LDS per WG: 16 hidden-state slots (16x128 bf16 = 4KB each) + q/m (4KB) + u (4KB) + att (1KB)
static constexpr int LDS_HS    = 16 * 16 * 128 * 2;            // 65536
static constexpr int LDS_TOTAL = LDS_HS + 4096 + 4096 + 1024;  // 74752

__device__ __forceinline__ f32x8 wmma_bf16(bf16x16 a, bf16x16 b, f32x8 c) {
  return __builtin_amdgcn_wmma_f32_16x16x32_bf16(false, a, false, b, (short)0, c, false, false);
}

__device__ __forceinline__ float fast_sigmoid(float x) {
  x = fminf(fmaxf(x, -30.f), 30.f);
  return 1.f / (1.f + __expf(-x));
}
__device__ __forceinline__ float fast_tanh(float x) {
  x = fminf(fmaxf(x, -15.f), 15.f);
  float e = __expf(2.f * x);
  return (e - 1.f) / (e + 1.f);
}

// 16x32 bf16 A-fragment chunk from a row-major [16][128] bf16 tile (CDNA5 16-bit A layout):
//   lane L: row = L&15, half = L>>4
//   elems 0..7  -> K = c*32 + half*8 + (0..7)
//   elems 8..15 -> K = c*32 + 16 + half*8 + (0..7)
__device__ __forceinline__ bf16x16 load_a_frag(const bf16_t* rowbase, int c, int half) {
  const bf16x8 lo = *(const bf16x8*)(rowbase + c * 32 + half * 8);
  const bf16x8 hi = *(const bf16x8*)(rowbase + c * 32 + 16 + half * 8);
  bf16x16 a;
#pragma unroll
  for (int i = 0; i < 8; ++i) { a[i] = lo[i]; a[8 + i] = hi[i]; }
  return a;
}

// ---------------- prep kernels ----------------

__global__ void k_f32_to_bf16(const float* __restrict__ src, bf16_t* __restrict__ dst, int n) {
  int i = blockIdx.x * blockDim.x + threadIdx.x;
  if (i < n) dst[i] = (bf16_t)src[i];
}

// Pre-swizzle a KxN f32 matrix into per-WMMA B-fragments (bf16):
//   tile (kc, nt): lane L holds 16 bf16: B[kc*32 + (e<8 ? (L>>4)*8 + e : 16 + (L>>4)*8 + e-8)][nt*16 + (L&15)]
// transpose=1 swizzles src^T.
__global__ void k_swizzle_b(const float* __restrict__ src, bf16_t* __restrict__ dst,
                            int K, int N, int transpose) {
  int tid = blockIdx.x * blockDim.x + threadIdx.x;
  int NT = N >> 4, KC = K >> 5;
  int total = KC * NT * 32;
  if (tid >= total) return;
  int lane = tid & 31;
  int tile = tid >> 5;
  int nt = tile % NT;
  int kc = tile / NT;
  int col = lane & 15;
  int half = lane >> 4;
  bf16_t* o = dst + (size_t)tile * 512 + lane * 16;
#pragma unroll
  for (int e = 0; e < 16; ++e) {
    int k = kc * 32 + (e < 8 ? half * 8 + e : 16 + half * 8 + (e - 8));
    int n = nt * 16 + col;
    float v = transpose ? src[(size_t)n * K + k] : src[(size_t)k * N + n];
    o[e] = (bf16_t)v;
  }
}

// ---------------- main fused GRU + attention kernel ----------------
// One workgroup (4 waves) per 16-path tile. Wave w owns N-chunks {2w, 2w+1} of every
// gate; its 48 GRU weight B-fragments are preloaded into registers before the
// 16-step recurrence, so the steady-state loop issues only WMMAs + LDS traffic.

__global__ void __launch_bounds__(128)
k_path_gru_att(const bf16_t* __restrict__ edges,      // [NUM_EDGES][128] bf16
               const long long* __restrict__ paths,   // [NUM_PATHS*16] int64
               const float* __restrict__ bias_i,      // [384]
               const float* __restrict__ bias_r,      // [384]
               const bf16_t* __restrict__ Wk_swz,     // kernel     (128x384), tiles kc*24+nt
               const bf16_t* __restrict__ Wr_swz,     // rec_kernel (128x384)
               const bf16_t* __restrict__ Wq_swz,     // wq   (128x128), tiles kc*8+nt
               const bf16_t* __restrict__ WkT_swz,    // wk^T (128x128)
               const bf16_t* __restrict__ Wv_swz,     // wv   (128x128)
               float* __restrict__ out,               // [NUM_PATHS][128]
               int num_tiles) {
  extern __shared__ char smem[];
  const int lane = threadIdx.x & 31;
  const int wave = threadIdx.x >> 5;
  const int tile = blockIdx.x;
  if (tile >= num_tiles) return;  // whole WG returns together
  const int col = lane & 15;
  const int half = lane >> 4;

  bf16_t* hs  = (bf16_t*)smem;                     // 16 slots x [16][128] bf16
  bf16_t* qm  = (bf16_t*)(smem + LDS_HS);          // q, later m  [16][128]
  bf16_t* uu  = (bf16_t*)(smem + LDS_HS + 4096);   // u           [16][128]
  float*  att = (float*)(smem + LDS_HS + 8192);    // [16][16]

  const int p0 = tile * 16;
  const long long* prow = paths + (size_t)(p0 + col) * MAX_LEN;

  // ---- preload this wave's GRU weight fragments (held across the whole recurrence) ----
  bf16x16 wX[3][2][4];  // [gate z/r/h][local n][kc]  from kernel
  bf16x16 wH[3][2][4];  //                             from rec_kernel
#pragma unroll
  for (int g = 0; g < 3; ++g)
#pragma unroll
    for (int i = 0; i < 2; ++i) {
      const int nb = wave * 2 + i;
#pragma unroll
      for (int c = 0; c < 4; ++c) {
        wX[g][i][c] = *(const bf16x16*)(Wk_swz + ((c * 24 + g * 8 + nb) * 512) + lane * 16);
        wH[g][i][c] = *(const bf16x16*)(Wr_swz + ((c * 24 + g * 8 + nb) * 512) + lane * 16);
      }
    }

  // biases for this wave's two N-chunks
  float bz[2], br[2], bih[2], brh[2];
#pragma unroll
  for (int i = 0; i < 2; ++i) {
    const int nc = (wave * 2 + i) * 16 + col;
    bz[i]  = bias_i[nc] + bias_r[nc];
    br[i]  = bias_i[128 + nc] + bias_r[128 + nc];
    bih[i] = bias_i[256 + nc];
    brh[i] = bias_r[256 + nc];
  }

  f32x8 hF[2];
#pragma unroll
  for (int i = 0; i < 2; ++i) hF[i] = (f32x8){0.f, 0.f, 0.f, 0.f, 0.f, 0.f, 0.f, 0.f};

  // ---------------- GRU over 16 steps ----------------
#pragma unroll 1
  for (int t = 0; t < MAX_LEN; ++t) {
    // gather x_t A-fragments (per-lane row = this lane's path); all waves load same rows (L2-hot)
    long long e = prow[t];
    const bf16_t* xrow = edges + (size_t)e * LINK_DIM;
    bf16x16 aX[4];
#pragma unroll
    for (int c = 0; c < 4; ++c) aX[c] = load_a_frag(xrow, c, half);

    bf16x16 aH[4];
    if (t > 0) {
      const bf16_t* hrow = hs + (size_t)(t - 1) * 2048 + col * 128;
#pragma unroll
      for (int c = 0; c < 4; ++c) aH[c] = load_a_frag(hrow, c, half);
    }

    bf16_t* hslot = hs + (size_t)t * 2048;

#pragma unroll
    for (int i = 0; i < 2; ++i) {
      f32x8 az  = (f32x8){0.f, 0.f, 0.f, 0.f, 0.f, 0.f, 0.f, 0.f};
      f32x8 ar  = az, axh = az, ahh = az;
#pragma unroll
      for (int c = 0; c < 4; ++c) {
        az  = wmma_bf16(aX[c], wX[0][i][c], az);
        ar  = wmma_bf16(aX[c], wX[1][i][c], ar);
        axh = wmma_bf16(aX[c], wX[2][i][c], axh);
      }
      if (t > 0) {
#pragma unroll
        for (int c = 0; c < 4; ++c) {
          az  = wmma_bf16(aH[c], wH[0][i][c], az);
          ar  = wmma_bf16(aH[c], wH[1][i][c], ar);
          ahh = wmma_bf16(aH[c], wH[2][i][c], ahh);
        }
      }
      const int nb = wave * 2 + i;
      f32x8 hn;
#pragma unroll
      for (int j = 0; j < 8; ++j) {
        float z = fast_sigmoid(az[j] + bz[i]);
        float r = fast_sigmoid(ar[j] + br[i]);
        float cand = fast_tanh(axh[j] + bih[i] + r * (ahh[j] + brh[i]));
        float h = z * hF[i][j] + (1.f - z) * cand;
        hn[j] = h;
        hslot[(j + 8 * half) * 128 + nb * 16 + col] = (bf16_t)h;  // D-layout -> row-major
      }
      hF[i] = hn;
    }
    __syncthreads();  // h_t complete for all 8 N-chunks before step t+1 reads it
  }

  // ---------------- attention ----------------
  // q = last_state @ Wq   (each wave: its 2 N-chunks)
  {
    const bf16_t* hlast = hs + 15 * 2048 + col * 128;
    bf16x16 aL[4];
#pragma unroll
    for (int c = 0; c < 4; ++c) aL[c] = load_a_frag(hlast, c, half);
#pragma unroll
    for (int i = 0; i < 2; ++i) {
      const int nb = wave * 2 + i;
      f32x8 acc = (f32x8){0.f, 0.f, 0.f, 0.f, 0.f, 0.f, 0.f, 0.f};
#pragma unroll
      for (int c = 0; c < 4; ++c)
        acc = wmma_bf16(aL[c], *(const bf16x16*)(Wq_swz + ((c * 8 + nb) * 512) + lane * 16), acc);
#pragma unroll
      for (int j = 0; j < 8; ++j)
        qm[(j + 8 * half) * 128 + nb * 16 + col] = (bf16_t)acc[j];
    }
  }
  __syncthreads();
  // u = q @ Wk^T   (so att[p,l] = h_l . u_p)
  {
    const bf16_t* qrow = qm + col * 128;
    bf16x16 aQ[4];
#pragma unroll
    for (int c = 0; c < 4; ++c) aQ[c] = load_a_frag(qrow, c, half);
#pragma unroll
    for (int i = 0; i < 2; ++i) {
      const int nb = wave * 2 + i;
      f32x8 acc = (f32x8){0.f, 0.f, 0.f, 0.f, 0.f, 0.f, 0.f, 0.f};
#pragma unroll
      for (int c = 0; c < 4; ++c)
        acc = wmma_bf16(aQ[c], *(const bf16x16*)(WkT_swz + ((c * 8 + nb) * 512) + lane * 16), acc);
#pragma unroll
      for (int j = 0; j < 8; ++j)
        uu[(j + 8 * half) * 128 + nb * 16 + col] = (bf16_t)acc[j];
    }
  }
  __syncthreads();
  // att[p,l] = h_l[p,:] . u[p,:]   thread (wave,lane): p = col, l = wave*4 + half*2 + {0,1}
  {
    const bf16_t* urow = uu + col * 128;
#pragma unroll
    for (int j = 0; j < 2; ++j) {
      int l = wave * 4 + half * 2 + j;
      const bf16_t* hrow = hs + (size_t)l * 2048 + col * 128;
      float s = 0.f;
#pragma unroll
      for (int kb = 0; kb < 16; ++kb) {
        bf16x8 hv = *(const bf16x8*)(hrow + kb * 8);
        bf16x8 uv = *(const bf16x8*)(urow + kb * 8);
#pragma unroll
        for (int ii = 0; ii < 8; ++ii) s += (float)hv[ii] * (float)uv[ii];
      }
      att[col * 16 + l] = s;
    }
  }
  __syncthreads();
  // m[p,:] = sum_l att[p,l] * h_l[p,:]   thread: p = col, k-block b = wave*2 + half (16 k's)
  {
    const int b = wave * 2 + half;
    float acc[16];
#pragma unroll
    for (int ii = 0; ii < 16; ++ii) acc[ii] = 0.f;
#pragma unroll
    for (int l = 0; l < 16; ++l) {
      float a = att[col * 16 + l];
      const bf16_t* hp = hs + (size_t)l * 2048 + col * 128 + b * 16;
      bf16x8 h0 = *(const bf16x8*)(hp);
      bf16x8 h1 = *(const bf16x8*)(hp + 8);
#pragma unroll
      for (int ii = 0; ii < 8; ++ii) { acc[ii] += a * (float)h0[ii]; acc[8 + ii] += a * (float)h1[ii]; }
    }
    bf16x8 m0, m1;
#pragma unroll
    for (int ii = 0; ii < 8; ++ii) { m0[ii] = (bf16_t)acc[ii]; m1[ii] = (bf16_t)acc[8 + ii]; }
    *(bf16x8*)(qm + col * 128 + b * 16) = m0;       // reuse q region for m
    *(bf16x8*)(qm + col * 128 + b * 16 + 8) = m1;
  }
  __syncthreads();
  // context = m @ Wv  -> out   (each wave: its 2 N-chunks)
  {
    const bf16_t* mrow = qm + col * 128;
    bf16x16 aM[4];
#pragma unroll
    for (int c = 0; c < 4; ++c) aM[c] = load_a_frag(mrow, c, half);
#pragma unroll
    for (int i = 0; i < 2; ++i) {
      const int nb = wave * 2 + i;
      f32x8 acc = (f32x8){0.f, 0.f, 0.f, 0.f, 0.f, 0.f, 0.f, 0.f};
#pragma unroll
      for (int c = 0; c < 4; ++c)
        acc = wmma_bf16(aM[c], *(const bf16x16*)(Wv_swz + ((c * 8 + nb) * 512) + lane * 16), acc);
#pragma unroll
      for (int j = 0; j < 8; ++j)
        out[(size_t)(p0 + j + 8 * half) * 128 + nb * 16 + col] = acc[j];
    }
  }
}

// ---------------- host launch ----------------

extern "C" void kernel_launch(void* const* d_in, const int* in_sizes, int n_in,
                              void* d_out, int out_size, void* d_ws, size_t ws_size,
                              hipStream_t stream) {
  const float*     inputs = (const float*)d_in[0];
  const long long* paths  = (const long long*)d_in[1];
  // d_in[2]=idx, d_in[3]=seqs: deterministic, unused
  const float* kern   = (const float*)d_in[4];
  const float* rec    = (const float*)d_in[5];
  const float* bias_i = (const float*)d_in[6];
  const float* bias_r = (const float*)d_in[7];
  const float* wq     = (const float*)d_in[8];
  const float* wk     = (const float*)d_in[9];
  const float* wv     = (const float*)d_in[10];

  char* ws = (char*)d_ws;
  size_t off = 0;
  bf16_t* edges_bf = (bf16_t*)(ws + off); off += (size_t)NUM_EDGES * LINK_DIM * 2;
  bf16_t* Wk_swz   = (bf16_t*)(ws + off); off += (size_t)128 * 384 * 2;
  bf16_t* Wr_swz   = (bf16_t*)(ws + off); off += (size_t)128 * 384 * 2;
  bf16_t* Wq_swz   = (bf16_t*)(ws + off); off += (size_t)128 * 128 * 2;
  bf16_t* WkT_swz  = (bf16_t*)(ws + off); off += (size_t)128 * 128 * 2;
  bf16_t* Wv_swz   = (bf16_t*)(ws + off); off += (size_t)128 * 128 * 2;

  int nE = NUM_EDGES * LINK_DIM;
  k_f32_to_bf16<<<(nE + 255) / 256, 256, 0, stream>>>(inputs, edges_bf, nE);
  k_swizzle_b<<<(4 * 24 * 32 + 255) / 256, 256, 0, stream>>>(kern, Wk_swz, 128, 384, 0);
  k_swizzle_b<<<(4 * 24 * 32 + 255) / 256, 256, 0, stream>>>(rec,  Wr_swz, 128, 384, 0);
  k_swizzle_b<<<(4 * 8 * 32 + 255) / 256, 256, 0, stream>>>(wq, Wq_swz, 128, 128, 0);
  k_swizzle_b<<<(4 * 8 * 32 + 255) / 256, 256, 0, stream>>>(wk, WkT_swz, 128, 128, 1);
  k_swizzle_b<<<(4 * 8 * 32 + 255) / 256, 256, 0, stream>>>(wv, Wv_swz, 128, 128, 0);

  int num_tiles = NUM_PATHS / 16;  // 1875 (exact)
  dim3 grid(num_tiles);
  k_path_gru_att<<<grid, 128, LDS_TOTAL, stream>>>(
      edges_bf, paths, bias_i, bias_r, Wk_swz, Wr_swz, Wq_swz, WkT_swz, Wv_swz,
      (float*)d_out, num_tiles);
}
